// PLinear_79671643341529
// MI455X (gfx1250) — compile-verified, matrix operand order checked by reading, not run on hardware
//
#include <hip/hip_runtime.h>

#define IN_F  4096
#define OUT_F 4096
#define N_TOK 8192

typedef __attribute__((ext_vector_type(16))) __bf16 v16bf;
typedef __attribute__((ext_vector_type(8)))  __bf16 v8bf;
typedef __attribute__((ext_vector_type(4)))  __bf16 v4bf;
typedef __attribute__((ext_vector_type(8)))  float  v8f;
typedef __attribute__((ext_vector_type(4)))  float  v4f;

// ---------------------------------------------------------------------------
// Split fp32 -> bf16 hi + bf16 residual (bf16x3 GEMM operands)
// ---------------------------------------------------------------------------
__device__ __forceinline__ void split_bf16(float w, __bf16& hi, __bf16& lo) {
    __bf16 h = (__bf16)w;          // RNE convert
    hi = h;
    lo = (__bf16)(w - (float)h);   // residual captures ~8 more mantissa bits
}

// w = mean * (1 + noise * |sigma| * sigma_mult), split into wh/wl
__global__ __launch_bounds__(256) void PLinear_split_w_kernel(
    const float* __restrict__ mean, const float* __restrict__ sigma,
    const float* __restrict__ noise, const int* __restrict__ smult,
    __bf16* __restrict__ wh, __bf16* __restrict__ wl, int n4)
{
    const float ms = (float)(*smult);
    int i = blockIdx.x * 256 + threadIdx.x;
    const int stride = gridDim.x * 256;
    for (; i < n4; i += stride) {
        v4f m  = ((const v4f*)mean)[i];
        v4f s  = ((const v4f*)sigma)[i];
        v4f nz = ((const v4f*)noise)[i];
        v4bf hv, lv;
#pragma unroll
        for (int j = 0; j < 4; ++j) {
            float w = m[j] * (1.0f + nz[j] * __builtin_fabsf(s[j]) * ms);
            __bf16 h, l; split_bf16(w, h, l);
            hv[j] = h; lv[j] = l;
        }
        ((v4bf*)wh)[i] = hv;
        ((v4bf*)wl)[i] = lv;
    }
}

__global__ __launch_bounds__(256) void PLinear_split_x_kernel(
    const float* __restrict__ x, __bf16* __restrict__ xh, __bf16* __restrict__ xl, int n4)
{
    int i = blockIdx.x * 256 + threadIdx.x;
    const int stride = gridDim.x * 256;
    for (; i < n4; i += stride) {
        v4f v = ((const v4f*)x)[i];
        v4bf hv, lv;
#pragma unroll
        for (int j = 0; j < 4; ++j) {
            __bf16 h, l; split_bf16(v[j], h, l);
            hv[j] = h; lv[j] = l;
        }
        ((v4bf*)xh)[i] = hv;
        ((v4bf*)xl)[i] = lv;
    }
}

// ---------------------------------------------------------------------------
// WMMA fragment gather from a pointer already positioned at
// (row, k + 8*(lane/16)). ISA 16-bit A/B layout: lane L holds row (L%16);
// VGPRs 0-3 hold K = 8*(L/16)+0..7, VGPRs 4-7 hold K = 16+8*(L/16)+0..7.
// Two contiguous 16-byte loads; +16 elements folds into IOFFSET.
// ---------------------------------------------------------------------------
__device__ __forceinline__ v16bf frag_from(const __bf16* p)
{
    v8bf a = *(const v8bf*)p;
    v8bf b = *(const v8bf*)(p + 16);
    v16bf f;
#pragma unroll
    for (int i = 0; i < 8; ++i) { f[i] = a[i]; f[i + 8] = b[i]; }
    return f;
}

// ---------------------------------------------------------------------------
// bf16x3 GEMM: out[t][o] = sum_k x[t][k] * w[o][k] + bias[o]
// Block: 256 threads = 8 waves (4 M x 2 N). Wave tile: 32(M) x 64(N)
// = 2x4 accumulators of 16x16 fp32. Block tile: 128 x 128.
// Addressing: uniform scalar base (xh/xl/wh/wl advanced by k) + constant
// 32-bit per-lane offsets -> GVS saddr form, no per-iteration VALU.
// ---------------------------------------------------------------------------
__global__ __launch_bounds__(256) void PLinear_wmma_kernel(
    const __bf16* __restrict__ xh, const __bf16* __restrict__ xl,
    const __bf16* __restrict__ wh, const __bf16* __restrict__ wl,
    const float* __restrict__ bias, float* __restrict__ out)
{
    const int lane = threadIdx.x & 31;
    const int wave = threadIdx.x >> 5;     // 0..7
    const int wm   = wave >> 1;            // 0..3 (M)
    const int wn   = wave & 1;             // 0..1 (N)
    const int m0   = blockIdx.y * 128 + wm * 32;   // 2 M-tiles: m0, m0+16
    const int n0   = blockIdx.x * 128 + wn * 64;   // 4 N-tiles: n0 .. n0+48

    const int row  = lane & 15;
    const int ksub = (lane >> 4) << 3;     // 0 or 8

    // Per-lane element offsets, constant across the K loop (32-bit).
    unsigned int offA[2], offB[4];
#pragma unroll
    for (int mi = 0; mi < 2; ++mi)
        offA[mi] = (unsigned int)(m0 + 16 * mi + row) * IN_F + ksub;
#pragma unroll
    for (int ni = 0; ni < 4; ++ni)
        offB[ni] = (unsigned int)(n0 + 16 * ni + row) * IN_F + ksub;

    v8f acc[2][4] = {};

#pragma unroll 2
    for (int k = 0; k < IN_F; k += 32) {
        const __bf16* xhk = xh + k;   // uniform (SGPR) bases
        const __bf16* xlk = xl + k;
        const __bf16* whk = wh + k;
        const __bf16* wlk = wl + k;

        v16bf ah[2], al[2], bh[4], bl[4];
#pragma unroll
        for (int mi = 0; mi < 2; ++mi) {
            ah[mi] = frag_from(xhk + offA[mi]);
            al[mi] = frag_from(xlk + offA[mi]);
        }
#pragma unroll
        for (int ni = 0; ni < 4; ++ni) {
            bh[ni] = frag_from(whk + offB[ni]);
            bl[ni] = frag_from(wlk + offB[ni]);
        }
#pragma unroll
        for (int mi = 0; mi < 2; ++mi) {
#pragma unroll
            for (int ni = 0; ni < 4; ++ni) {
                // bf16x3: hi*hi + hi*lo + lo*hi (lo*lo ~2^-18, dropped)
                acc[mi][ni] = __builtin_amdgcn_wmma_f32_16x16x32_bf16(
                    false, ah[mi], false, bh[ni], (short)0, acc[mi][ni], false, false);
                acc[mi][ni] = __builtin_amdgcn_wmma_f32_16x16x32_bf16(
                    false, ah[mi], false, bl[ni], (short)0, acc[mi][ni], false, false);
                acc[mi][ni] = __builtin_amdgcn_wmma_f32_16x16x32_bf16(
                    false, al[mi], false, bh[ni], (short)0, acc[mi][ni], false, false);
            }
        }
    }

    // D layout: VGPR r -> m = r + 8*(lane/16); n = lane%16
    const int col  = lane & 15;
    const int rsel = (lane >> 4) << 3;
#pragma unroll
    for (int ni = 0; ni < 4; ++ni) {
        const int n = n0 + 16 * ni + col;
        const float b = bias[n];
#pragma unroll
        for (int mi = 0; mi < 2; ++mi) {
            const unsigned int obase = (unsigned int)(m0 + 16 * mi + rsel) * OUT_F + n;
#pragma unroll
            for (int r = 0; r < 8; ++r)
                out[obase + (unsigned int)r * OUT_F] = acc[mi][ni][r] + b;
        }
    }
}

// ---------------------------------------------------------------------------
// Launch
// ---------------------------------------------------------------------------
extern "C" void kernel_launch(void* const* d_in, const int* in_sizes, int n_in,
                              void* d_out, int out_size, void* d_ws, size_t ws_size,
                              hipStream_t stream) {
    const float* x     = (const float*)d_in[0];
    const float* mean  = (const float*)d_in[1];
    const float* sigma = (const float*)d_in[2];
    const float* bias  = (const float*)d_in[3];
    const float* noise = (const float*)d_in[4];
    const int*   smult = (const int*)d_in[5];
    float* out = (float*)d_out;

    // Workspace layout: wh(32MB) | wl(32MB) | xh(64MB) | xl(64MB) = 192MB
    __bf16* wh = (__bf16*)d_ws;
    __bf16* wl = wh + (size_t)OUT_F * IN_F;
    __bf16* xh = wl + (size_t)OUT_F * IN_F;
    __bf16* xl = xh + (size_t)N_TOK * IN_F;

    const int nw4 = (OUT_F * IN_F) / 4;   // 4.19M vec4 elems
    const int nx4 = (N_TOK * IN_F) / 4;   // 8.39M vec4 elems

    PLinear_split_w_kernel<<<2048, 256, 0, stream>>>(mean, sigma, noise, smult, wh, wl, nw4);
    PLinear_split_x_kernel<<<4096, 256, 0, stream>>>(x, xh, xl, nx4);

    dim3 grid(OUT_F / 128, N_TOK / 128);  // (32, 64)
    PLinear_wmma_kernel<<<grid, 256, 0, stream>>>(xh, xl, wh, wl, bias, out);
}
